// RingFlashCrossAttention_56495999811612
// MI455X (gfx1250) — matrix operand order repaired and checked by reading, not compiled, not run on hardware
//
#include <hip/hip_runtime.h>

typedef __attribute__((ext_vector_type(16))) __bf16 v16bf;
typedef __attribute__((ext_vector_type(8)))  __bf16 v8bf;
typedef __attribute__((ext_vector_type(8)))  float  v8f;

#define B_   4
#define SQ_  2048
#define SK_  2048
#define NH_  16
#define DH_  128
#define QT_  128     // Q rows per workgroup (8 waves x 16)
#define KT_  64      // keys per SK tile
#define KROW 136     // K LDS row stride in bf16 (128 + 8 pad -> 4-bank shift/row)
#define VROW 72      // V^T LDS row stride in bf16 (64 + 8 pad -> 36-bank shift/row)
#define PROW 72      // P LDS row stride in bf16

static __device__ __forceinline__ v16bf ld_cat(const __bf16* p0, const __bf16* p1) {
  v8bf a = *(const v8bf*)p0;
  v8bf b = *(const v8bf*)p1;
  return __builtin_shufflevector(a, b, 0,1,2,3,4,5,6,7,8,9,10,11,12,13,14,15);
}

// ---- DPP16 cross-lane reductions within each 16-lane half (no LDS traffic) ----
#define DPP_XOR1 0xB1   // quad_perm [1,0,3,2]
#define DPP_XOR2 0x4E   // quad_perm [2,3,0,1]
#define DPP_HMIR 0x141  // row_half_mirror (combines quad pairs)
#define DPP_MIR  0x140  // row_mirror      (combines 8-lane halves)

template <int CTRL>
static __device__ __forceinline__ float dpp_mov(float x) {
  int xi = __builtin_bit_cast(int, x);
  int r = __builtin_amdgcn_update_dpp(xi, xi, CTRL, 0xf, 0xf, true);
  return __builtin_bit_cast(float, r);
}
static __device__ __forceinline__ float red16_max(float x) {
  x = fmaxf(x, dpp_mov<DPP_XOR1>(x));
  x = fmaxf(x, dpp_mov<DPP_XOR2>(x));
  x = fmaxf(x, dpp_mov<DPP_HMIR>(x));
  x = fmaxf(x, dpp_mov<DPP_MIR>(x));
  return x;
}
static __device__ __forceinline__ float red16_sum(float x) {
  x += dpp_mov<DPP_XOR1>(x);
  x += dpp_mov<DPP_XOR2>(x);
  x += dpp_mov<DPP_HMIR>(x);
  x += dpp_mov<DPP_MIR>(x);
  return x;
}

__global__ __launch_bounds__(256) void fa_fwd_kernel(const float* __restrict__ q,
                                                     const float* __restrict__ kv,
                                                     float* __restrict__ out)
{
  __shared__ __align__(16) __bf16 Kl[KT_ * KROW];       // K tile, [key][d]
  __shared__ __align__(16) __bf16 Vt[DH_ * VROW];       // V tile transposed, [d][key^swz]
  __shared__ __align__(16) __bf16 Pl[8 * 16 * PROW];    // per-wave P scratch

  const int b    = blockIdx.z;
  const int h    = blockIdx.y;
  const int q0   = blockIdx.x * QT_;
  const int tid  = threadIdx.x;
  const int wave = tid >> 5;
  const int lane = tid & 31;
  const int lm   = lane & 15;   // A: row M / B,C: col N
  const int lh   = lane >> 4;   // half selector

  // 1/sqrt(128) * log2(e): fold softmax scale + base-2 exp into Q
  const float qscale = 0.08838834764831845f * 1.4426950408889634f;

  // ---- Q fragments in A layout (16x32 bf16 per 32-wide D chunk) ----
  v16bf qa[4];
  {
    const int row = q0 + wave * 16 + lm;
    const float* qp = q + (((size_t)b * SQ_ + row) * NH_ + h) * DH_;
    #pragma unroll
    for (int f = 0; f < 4; ++f) {
      float t[16];
      *(float4*)&t[0]  = *(const float4*)(qp + 32*f + 8*lh);
      *(float4*)&t[4]  = *(const float4*)(qp + 32*f + 8*lh + 4);
      *(float4*)&t[8]  = *(const float4*)(qp + 32*f + 8*lh + 16);
      *(float4*)&t[12] = *(const float4*)(qp + 32*f + 8*lh + 20);
      #pragma unroll
      for (int j = 0; j < 16; ++j) qa[f][j] = (__bf16)(t[j] * qscale);
    }
  }

  v8f acc[8] = {};              // O accumulators: 8 d-groups x 8 rows/lane
  float mrow[8], lrow[8];
  #pragma unroll
  for (int r = 0; r < 8; ++r) { mrow[r] = -3.0e38f; lrow[r] = 0.0f; }

  const int key_cp = tid >> 2;          // 0..63: key this thread copies
  const int d_cp   = (tid & 3) * 32;    // 0,32,64,96: 32-float d-chunk
  const int vswz   = key_cp ^ (8 * (tid & 3));  // store-side column swizzle (constant/thread)

  for (int s0 = 0; s0 < SK_; s0 += KT_) {
    __syncthreads();   // previous tile's fragment reads done before overwrite
    {
      const float* kp = kv + ((((size_t)b * SK_ + (s0 + key_cp)) * 2) * NH_ + h) * DH_ + d_cp;
      const float* vp = kp + (size_t)NH_ * DH_;   // c=0 -> c=1
      float kt[32];
      #pragma unroll
      for (int j = 0; j < 8; ++j) *(float4*)&kt[4*j] = *(const float4*)(kp + 4*j);
      #pragma unroll
      for (int jj = 0; jj < 4; ++jj) {
        v8bf kb;
        #pragma unroll
        for (int j = 0; j < 8; ++j) kb[j] = (__bf16)kt[8*jj + j];
        *(v8bf*)&Kl[key_cp * KROW + d_cp + 8*jj] = kb;
      }
      float vt[32];
      #pragma unroll
      for (int j = 0; j < 8; ++j) *(float4*)&vt[4*j] = *(const float4*)(vp + 4*j);
      #pragma unroll
      for (int j = 0; j < 32; ++j) Vt[(d_cp + j) * VROW + vswz] = (__bf16)vt[j];
    }
    if (s0 + KT_ < SK_) {   // prefetch next tile into the cache hierarchy
      const float* np = kv + ((((size_t)b * SK_ + (s0 + KT_ + key_cp)) * 2) * NH_ + h) * DH_ + d_cp;
      __builtin_prefetch(np, 0, 1);
      __builtin_prefetch(np + (size_t)NH_ * DH_, 0, 1);
    }
    __syncthreads();

    // ---- S = Q K^T : four 16x16 column groups, 4 WMMAs each over D=128 ----
    v8f sfr[4] = {};
    #pragma unroll
    for (int f = 0; f < 4; ++f) {
      #pragma unroll
      for (int t2 = 0; t2 < 4; ++t2) {
        const __bf16* kp0 = &Kl[(16 * t2 + lm) * KROW + 32 * f + 8 * lh];
        v16bf kb = ld_cat(kp0, kp0 + 16);
        sfr[t2] = __builtin_amdgcn_wmma_f32_16x16x32_bf16(
            false, qa[f], false, kb, (short)0, sfr[t2], false, false);
      }
    }

    // ---- online softmax: pure-VALU DPP reductions across the 16-lane halves ----
    #pragma unroll
    for (int r = 0; r < 8; ++r) {
      float t = fmaxf(fmaxf(sfr[0][r], sfr[1][r]), fmaxf(sfr[2][r], sfr[3][r]));
      t = red16_max(t);
      const float mnew = fmaxf(mrow[r], t);
      const float corr = exp2f(mrow[r] - mnew);
      mrow[r] = mnew;
      float rs = 0.0f;
      #pragma unroll
      for (int t2 = 0; t2 < 4; ++t2) {
        const float p = exp2f(sfr[t2][r] - mnew);
        sfr[t2][r] = p;
        rs += p;
      }
      rs = red16_sum(rs);
      lrow[r] = lrow[r] * corr + rs;
      #pragma unroll
      for (int g = 0; g < 8; ++g) acc[g][r] *= corr;
    }

    // ---- re-layout P (C-frag -> A-frag) via per-wave LDS scratch ----
    const int pbase = wave * 16 * PROW;
    #pragma unroll
    for (int t2 = 0; t2 < 4; ++t2)
      #pragma unroll
      for (int r = 0; r < 8; ++r)
        Pl[pbase + (8 * lh + r) * PROW + 16 * t2 + lm] = (__bf16)sfr[t2][r];

    v16bf pa[2];
    #pragma unroll
    for (int c = 0; c < 2; ++c) {
      const __bf16* pp = &Pl[pbase + lm * PROW + 32 * c + 8 * lh];
      pa[c] = ld_cat(pp, pp + 16);
    }

    // ---- O += P V : 8 d-groups x 2 key-chunks ----
    #pragma unroll
    for (int g = 0; g < 8; ++g) {
      const int sv = 8 * ((g >> 1) & 3);   // read-side swizzle (compile-time per g)
      #pragma unroll
      for (int c = 0; c < 2; ++c) {
        const __bf16* vp0 = &Vt[(16 * g + lm) * VROW + ((32 * c + 8 * lh) ^ sv)];
        v16bf vb = ld_cat(vp0, &Vt[(16 * g + lm) * VROW + ((32 * c + 8 * lh + 16) ^ sv)]);
        acc[g] = __builtin_amdgcn_wmma_f32_16x16x32_bf16(
            false, pa[c], false, vb, (short)0, acc[g], false, false);
      }
    }
  }

  // ---- epilogue: O /= l, write out ----
  {
    const int row0 = q0 + wave * 16 + 8 * lh;
    #pragma unroll
    for (int r = 0; r < 8; ++r) {
      const float inv = 1.0f / lrow[r];
      float* op = out + (((size_t)b * SQ_ + (row0 + r)) * NH_ + h) * DH_ + lm;
      #pragma unroll
      for (int g = 0; g < 8; ++g) op[16 * g] = acc[g][r] * inv;
    }
  }
}

extern "C" void kernel_launch(void* const* d_in, const int* in_sizes, int n_in,
                              void* d_out, int out_size, void* d_ws, size_t ws_size,
                              hipStream_t stream) {
  const float* q  = (const float*)d_in[0];
  const float* kv = (const float*)d_in[1];
  float* out = (float*)d_out;
  dim3 grid(SQ_ / QT_, NH_, B_);   // (16, 16, 4) = 1024 workgroups
  fa_fwd_kernel<<<grid, dim3(256, 1, 1), 0, stream>>>(q, kv, out);
}
